// LinearLocalAttention_6717328851261
// MI455X (gfx1250) — matrix operand order repaired and checked by reading, not compiled, not run on hardware
//
#include <hip/hip_runtime.h>
#include <hip/hip_bf16.h>

// ---------------------------------------------------------------------------
// Problem constants (from reference): B=2, N=4096, DIM=1024, H=8, HD=128,
// FEAT=64, WS=7.  M = B*N = 8192 rows for all big GEMMs.
// ---------------------------------------------------------------------------
#define CB   2
#define CN   4096
#define CDIM 1024
#define CH   8
#define CHD  128
#define CF   64

typedef __attribute__((ext_vector_type(16))) __bf16 v16bf;
typedef __attribute__((ext_vector_type(8)))  float  v8f;

union Frag16 { v16bf v; unsigned u[8]; };

__device__ __forceinline__ unsigned f2bf_bits(float f) {
    union { float f; unsigned u; } a; a.f = f;
    unsigned r = a.u + 0x7fffu + ((a.u >> 16) & 1u);   // round-to-nearest-even
    return r >> 16;
}
__device__ __forceinline__ __bf16 f2bf(float f) {
    unsigned short h = (unsigned short)f2bf_bits(f);
    __bf16 b; __builtin_memcpy(&b, &h, 2);
    return b;
}
__device__ __forceinline__ unsigned pack2bf(float lo, float hi) {
    return f2bf_bits(lo) | (f2bf_bits(hi) << 16);
}

__device__ __forceinline__ v8f wmma_bf16(v16bf a, v16bf b, v8f c) {
    return __builtin_amdgcn_wmma_f32_16x16x32_bf16(
        /*neg_a=*/false, a, /*neg_b=*/false, b,
        /*c_mod=*/(short)0, c, /*reuse_a=*/false, /*reuse_b=*/false);
}

// CDNA5 async global->LDS copy (ASYNCcnt-tracked), issued per-lane: 16B
// contiguous global -> 16B contiguous LDS.  vdst = 32-bit LDS byte address
// (low 32 bits of the generic pointer ARE the LDS offset per the aperture
// rules), vaddr = 64-bit global address.
__device__ __forceinline__ void async_load_b128(unsigned lds_addr, const void* gptr) {
    asm volatile("global_load_async_to_lds_b128 %0, %1, off"
                 :: "v"(lds_addr), "v"(gptr) : "memory");
}
__device__ __forceinline__ void wait_asynccnt0() {
    asm volatile("s_wait_asynccnt 0x0" ::: "memory");
}

// ---------------------------------------------------------------------------
// fp32 -> bf16 conversion, 4 elements per thread (float4 -> packed uint2)
// ---------------------------------------------------------------------------
__global__ void cvt_f32_bf16(const float4* __restrict__ in,
                             uint2* __restrict__ out, long n4) {
    long i = (long)blockIdx.x * blockDim.x + threadIdx.x;
    long stride = (long)gridDim.x * blockDim.x;
    for (; i < n4; i += stride) {
        float4 v = in[i];
        out[i] = make_uint2(pack2bf(v.x, v.y), pack2bf(v.z, v.w));
    }
}

// ---------------------------------------------------------------------------
// Tiled WMMA bf16 GEMM:  C[M,N] = A[M,K] (row-major bf16) @ B[K,N] (row-major
// bf16) (+ bias[N]).  Block = 256 threads = 8 waves; tile 128x128x64, LDS
// double-buffered.  A tile is filled by the CDNA5 async copy engine while the
// wave computes; B tile is prefetched to registers (two adjacent k-rows per
// thread) and stored transposed as packed (k,k+1) dwords.  Rows of both LDS
// tiles are padded to 72 elements (144B): 16B-aligned so fragment gathers
// merge into ds_load_b128, and the 36-dword stride spreads the 16 fragment
// rows over all 64 banks conflict-free.  Wave (wm,wn) in 2x4 grid, each wave
// owns a 64x32 C sub-tile (4x2 accums): 16 WMMAs per iteration, one barrier.
// ---------------------------------------------------------------------------
__global__ __launch_bounds__(256) void gemm_bf16_kernel(
    const __bf16* __restrict__ A, const __bf16* __restrict__ Bm,
    float* __restrict__ C, const float* __restrict__ bias,
    int M, int Nn, int K)
{
    constexpr int BM = 128, BN = 128, BK = 64;
    constexpr int PAD = 8;                    // row stride 72 elems = 144B
    __shared__ __bf16 As[2][BM][BK + PAD];    // [buf][m][k]
    __shared__ __bf16 Bs[2][BN][BK + PAD];    // [buf][n][k] (transposed fill)

    const int tid  = threadIdx.x;
    const int lane = tid & 31;
    const int wv   = tid >> 5;
    const int wm   = wv & 1;    // 0..1  (64 rows each)
    const int wn   = wv >> 1;   // 0..3  (32 cols each)
    const int m0   = blockIdx.y * BM;
    const int n0   = blockIdx.x * BN;
    const int hlf  = lane >> 4;       // lane half (0/1)
    const int l16  = lane & 15;

    v8f acc[4][2] = {};
    uint4 bregA[2], bregB[2];         // two adjacent k-rows, 8 n each

    auto issue_asyncA = [&](int buf, int kb) {
        #pragma unroll
        for (int t = 0; t < 4; ++t) {
            int s   = tid + t * 256;          // 1024 slots of 8 bf16
            int row = s >> 3;                 // 0..127
            int cc  = s & 7;                  // chunk of 8 bf16 (16B)
            const void* g = A + (long)(m0 + row) * K + kb + cc * 8;
            unsigned l = (unsigned)(unsigned long long)&As[buf][row][cc * 8];
            async_load_b128(l, g);
        }
    };
    auto loadB_regs = [&](int kb) {
        #pragma unroll
        for (int t = 0; t < 2; ++t) {
            int s  = tid + t * 256;           // 512 (k-pair, n-chunk) slots
            int kp = s >> 4;                  // 0..31 -> k = 2kp, 2kp+1
            int nc = s & 15;                  // chunk of 8 n
            const __bf16* base = Bm + (long)(kb + 2 * kp) * Nn + n0 + nc * 8;
            bregA[t] = *(const uint4*)(base);
            bregB[t] = *(const uint4*)(base + Nn);
        }
    };
    auto storeB = [&](int buf) {
        #pragma unroll
        for (int t = 0; t < 2; ++t) {
            int s  = tid + t * 256;
            int kp = s >> 4;
            int nc = s & 15;
            unsigned lo[4] = { bregA[t].x, bregA[t].y, bregA[t].z, bregA[t].w };
            unsigned hi[4] = { bregB[t].x, bregB[t].y, bregB[t].z, bregB[t].w };
            #pragma unroll
            for (int e = 0; e < 8; ++e) {
                unsigned l16v = (e & 1) ? (lo[e >> 1] >> 16) : (lo[e >> 1] & 0xffffu);
                unsigned h16v = (e & 1) ? (hi[e >> 1] >> 16) : (hi[e >> 1] & 0xffffu);
                *(unsigned*)&Bs[buf][nc * 8 + e][2 * kp] = l16v | (h16v << 16);
            }
        }
    };
    auto compute = [&](int buf) {
        #pragma unroll
        for (int ks = 0; ks < 2; ++ks) {      // two K=32 steps per tile
            const int kb32 = ks * 32;
            Frag16 af[4];
            #pragma unroll
            for (int i = 0; i < 4; ++i) {
                int row = wm * 64 + i * 16 + l16;
                #pragma unroll
                for (int v = 0; v < 8; ++v) {
                    int k = kb32 + ((v >> 2) << 4) + (hlf << 3) + ((v & 3) << 1);
                    af[i].u[v] = *(const unsigned*)&As[buf][row][k];
                }
            }
            Frag16 bfr[2];
            #pragma unroll
            for (int j = 0; j < 2; ++j) {
                int n = wn * 32 + j * 16 + l16;
                #pragma unroll
                for (int v = 0; v < 8; ++v) {
                    int k = kb32 + ((v >> 2) << 4) + (hlf << 3) + ((v & 3) << 1);
                    bfr[j].u[v] = *(const unsigned*)&Bs[buf][n][k];
                }
            }
            #pragma unroll
            for (int i = 0; i < 4; ++i)
                #pragma unroll
                for (int j = 0; j < 2; ++j)
                    acc[i][j] = wmma_bf16(af[i].v, bfr[j].v, acc[i][j]);
        }
    };

    // ---- prologue: fill buffer 0 ----
    issue_asyncA(0, 0);
    loadB_regs(0);
    storeB(0);
    wait_asynccnt0();
    __syncthreads();

    int cur = 0;
    for (int kb = 0; kb < K; kb += BK) {
        const int kn = kb + BK;
        const bool has_next = kn < K;
        if (has_next) {                 // overlap next-tile fill with WMMAs
            issue_asyncA(cur ^ 1, kn);  // async engine writes other buffer
            loadB_regs(kn);
        }
        compute(cur);
        if (has_next) {
            storeB(cur ^ 1);
            wait_asynccnt0();           // our async LDS writes complete
            __syncthreads();            // publish buffer to all waves
            cur ^= 1;
        }
    }

    // ---- epilogue: D layout n=lane%16, m=r+8*(lane/16) ----
    #pragma unroll
    for (int i = 0; i < 4; ++i)
        #pragma unroll
        for (int j = 0; j < 2; ++j) {
            int n = n0 + wn * 32 + j * 16 + l16;
            float bv = bias ? bias[n] : 0.0f;
            #pragma unroll
            for (int r = 0; r < 8; ++r) {
                int m = m0 + wm * 64 + i * 16 + r + (hlf << 3);
                C[(long)m * Nn + n] = acc[i][j][r] + bv;
            }
        }
}

// ---------------------------------------------------------------------------
// Feature map:  phi = elu( Q_h @ proj ) + 1   per (b,h).
// Q_h rows come from fp32 qkv (stride 3*DIM, col base col_off + h*128),
// converted to bf16 into LDS.  proj (128x64 fp32) converted once into LDS,
// stored [f][k].  Block = 128 threads = 4 waves, 64 rows x 64 feats per block.
// elu(x)+1 = x>0 ? x+1 : exp(x).
// ---------------------------------------------------------------------------
__global__ __launch_bounds__(128) void phi_kernel(
    const float* __restrict__ qkv, const float* __restrict__ proj,
    float* __restrict__ phi, int col_off)
{
    __shared__ __bf16 Ps[CF][CHD + 8];    // [f][k], 16B-aligned rows
    __shared__ __bf16 As[64][32 + 8];     // [m][k]

    const int tid  = threadIdx.x;
    const int lane = tid & 31;
    const int wv   = tid >> 5;           // 0..3 -> 16 rows each
    const int hlf  = lane >> 4;
    const int l16  = lane & 15;
    const int bh   = blockIdx.y;         // b*H + h
    const int b    = bh >> 3;
    const int h    = bh & 7;
    const long n0  = (long)blockIdx.x * 64;
    const int cbase = col_off + h * CHD;

    // proj -> Ps (8192 elems / 128 threads), coalesced on f
    for (int t = 0; t < 64; ++t) {
        int idx = tid + t * 128;
        int k = idx >> 6, f = idx & 63;
        Ps[f][k] = f2bf(proj[k * CF + f]);
    }

    v8f acc[4] = {};
    for (int kk = 0; kk < CHD; kk += 32) {
        __syncthreads();
        // qkv slice -> As (2048 elems / 128 threads), coalesced on k
        for (int t = 0; t < 16; ++t) {
            int idx = tid + t * 128;
            int m = idx >> 5, k = idx & 31;
            As[m][k] = f2bf(qkv[(long)(b * CN + n0 + m) * (3 * CDIM)
                                + cbase + kk + k]);
        }
        __syncthreads();

        Frag16 af;
        int row = wv * 16 + l16;
        #pragma unroll
        for (int v = 0; v < 8; ++v) {
            int k = ((v >> 2) << 4) + (hlf << 3) + ((v & 3) << 1);
            af.u[v] = *(const unsigned*)&As[row][k];
        }
        #pragma unroll
        for (int j = 0; j < 4; ++j) {
            Frag16 bf;
            int f = j * 16 + l16;
            #pragma unroll
            for (int v = 0; v < 8; ++v) {
                int k = kk + ((v >> 2) << 4) + (hlf << 3) + ((v & 3) << 1);
                bf.u[v] = *(const unsigned*)&Ps[f][k];
            }
            acc[j] = wmma_bf16(af.v, bf.v, acc[j]);
        }
    }

    #pragma unroll
    for (int j = 0; j < 4; ++j) {
        int f = j * 16 + l16;
        #pragma unroll
        for (int r = 0; r < 8; ++r) {
            long n = n0 + wv * 16 + r + (hlf << 3);
            float x = acc[j][r];
            float o = (x > 0.0f) ? (x + 1.0f) : __expf(x);
            phi[((long)(b * CH + h) * CN + n) * CF + f] = o;
        }
    }
}

// ---------------------------------------------------------------------------
// Window-7 linear local attention.  One wave32 per (b,h,n), float2 per lane:
//   lane covers f = 2*lane, 2*lane+1 for the s-dot (full shfl reduce),
//   and d = {2*lane, 2*lane+1, 64+2*lane, 65+2*lane} for the v-accumulate.
// Output written as packed 2xbf16 dwords in (b, n, h*128+d) layout.
// ---------------------------------------------------------------------------
__global__ __launch_bounds__(256) void local_attn_kernel(
    const float* __restrict__ qkv, const float* __restrict__ q_phi,
    const float* __restrict__ k_phi, __bf16* __restrict__ attn_out)
{
    const long gwave = ((long)blockIdx.x * blockDim.x + threadIdx.x) >> 5;
    const int lane   = threadIdx.x & 31;
    const int n  = (int)(gwave & (CN - 1));
    const int bh = (int)(gwave >> 12);           // N = 4096 = 2^12
    const int h  = bh & 7;
    const int b  = bh >> 3;

    const float2 q01 = ((const float2*)(q_phi
                        + ((long)(b * CH + h) * CN + n) * CF))[lane];

    float2 a0 = {0.f, 0.f}, a1 = {0.f, 0.f};
    float zsum = 0.f;
    #pragma unroll
    for (int w = 0; w < 7; ++w) {
        int pos = n + w - 3;                     // uniform per wave
        if (pos < 0 || pos >= CN) continue;
        const float2 k01 = ((const float2*)(k_phi
                            + ((long)(b * CH + h) * CN + pos) * CF))[lane];
        float s = q01.x * k01.x + q01.y * k01.y;
        #pragma unroll
        for (int off = 16; off > 0; off >>= 1) s += __shfl_xor(s, off, 32);
        zsum += s;
        const float2* vp = (const float2*)(qkv
                            + (long)(b * CN + pos) * (3 * CDIM)
                            + 2 * CDIM + h * CHD);
        float2 v0 = vp[lane], v1 = vp[lane + 32];
        a0.x += s * v0.x;  a0.y += s * v0.y;
        a1.x += s * v1.x;  a1.y += s * v1.y;
    }
    const float z = 1.0f / (zsum + 1e-8f);
    unsigned* op = (unsigned*)(attn_out + (long)(b * CN + n) * CDIM + h * CHD);
    op[lane]      = pack2bf(a0.x * z, a0.y * z);
    op[lane + 32] = pack2bf(a1.x * z, a1.y * z);
}

// ---------------------------------------------------------------------------
// Host-side orchestration
// ---------------------------------------------------------------------------
extern "C" void kernel_launch(void* const* d_in, const int* in_sizes, int n_in,
                              void* d_out, int out_size, void* d_ws, size_t ws_size,
                              hipStream_t stream) {
    (void)in_sizes; (void)n_in; (void)out_size; (void)ws_size;
    const float* x     = (const float*)d_in[0];   // (B,N,DIM)
    const float* w_qkv = (const float*)d_in[1];   // (DIM, 3*DIM)
    const float* proj  = (const float*)d_in[2];   // (HD, FEAT)
    const float* w_out = (const float*)d_in[3];   // (DIM, DIM)
    const float* b_out = (const float*)d_in[4];   // (DIM,)
    float* out = (float*)d_out;

    char* ws = (char*)d_ws;                       // 256B-aligned regions
    __bf16* x_bf    = (__bf16*)(ws + 0L);          // 8192*1024 bf16   (16 MB)
    __bf16* wqkv_bf = (__bf16*)(ws + 16777216L);   // 1024*3072 bf16   ( 6 MB)
    float*  qkv     = (float *)(ws + 23068672L);   // 8192*3072 f32    (96 MB)
    float*  q_phi   = (float *)(ws + 123731968L);  // 16*4096*64 f32   (16 MB)
    float*  k_phi   = (float *)(ws + 140509184L);  // 16*4096*64 f32   (16 MB)
    __bf16* attn_bf = (__bf16*)(ws + 157286400L);  // 8192*1024 bf16   (16 MB)
    __bf16* wout_bf = (__bf16*)(ws + 174063616L);  // 1024*1024 bf16   ( 2 MB)

    const long M = (long)CB * CN;                  // 8192

    cvt_f32_bf16<<<1024, 256, 0, stream>>>((const float4*)x,
                                           (uint2*)x_bf,    M * CDIM / 4);
    cvt_f32_bf16<<<1024, 256, 0, stream>>>((const float4*)w_qkv,
                                           (uint2*)wqkv_bf, (long)CDIM * 3 * CDIM / 4);
    cvt_f32_bf16<<<512, 256, 0, stream>>>((const float4*)w_out,
                                          (uint2*)wout_bf,  (long)CDIM * CDIM / 4);

    // QKV projection: (8192 x 1024) @ (1024 x 3072)
    gemm_bf16_kernel<<<dim3(3 * CDIM / 128, M / 128), 256, 0, stream>>>(
        x_bf, wqkv_bf, qkv, nullptr, (int)M, 3 * CDIM, CDIM);

    // Per-head feature maps with elu+1 epilogue
    phi_kernel<<<dim3(CN / 64, CB * CH), 128, 0, stream>>>(qkv, proj, q_phi, 0);
    phi_kernel<<<dim3(CN / 64, CB * CH), 128, 0, stream>>>(qkv, proj, k_phi, CDIM);

    // Window-7 local attention: 65536 waves, 8 per block
    local_attn_kernel<<<(CB * CH * CN) / 8, 256, 0, stream>>>(
        qkv, q_phi, k_phi, attn_bf);

    // Output projection + bias: (8192 x 1024) @ (1024 x 1024) + b_out
    gemm_bf16_kernel<<<dim3(CDIM / 128, M / 128), 256, 0, stream>>>(
        attn_bf, wout_bf, out, b_out, (int)M, CDIM, CDIM);
}
